// STPIGNN_38027640439389
// MI455X (gfx1250) — matrix operand order repaired and checked by reading, not compiled, not run on hardware
//
#include <hip/hip_runtime.h>

typedef __attribute__((ext_vector_type(16))) _Float16 v16h;
typedef __attribute__((ext_vector_type(8)))  _Float16 v8h;
typedef __attribute__((ext_vector_type(8)))  float    v8f;

#define BDIM 256

// Problem constants (from reference setup_inputs)
#define Bsz 2
#define Tt  8
#define Nn  10000
#define Ff  16
#define Ee  320000
#define Dd  16
#define Hh  128
#define Ll  2
#define BN  (Bsz * Nn)

// ---------------------------------------------------------------------------
// Weight packing: transpose to [out_col][k] f16 so each lane's WMMA B-fragment
// is two 16B reads within one row. K=16 mats are zero-padded to K=32.
// ---------------------------------------------------------------------------
__global__ void pack_weights(const float* __restrict__ enc_w, const float* __restrict__ eW,
                             const float* __restrict__ m1w, const float* __restrict__ m2w,
                             const float* __restrict__ wih, const float* __restrict__ whh,
                             _Float16* __restrict__ enc_wT, _Float16* __restrict__ eWT,
                             _Float16* __restrict__ m1wT, _Float16* __restrict__ m2wT,
                             _Float16* __restrict__ wihT, _Float16* __restrict__ whhT) {
    int idx = blockIdx.x * blockDim.x + threadIdx.x;
    switch (blockIdx.y) {
    case 0: // enc_wT [128][32], pad k>=16 with 0
        if (idx < 128 * 32) {
            int col = idx >> 5, k = idx & 31;
            enc_wT[idx] = (_Float16)((k < Ff) ? enc_w[k * Hh + col] : 0.f);
        }
        break;
    case 1: // eWT [2][128][32]
        if (idx < 2 * 128 * 32) {
            int l = idx >> 12, col = (idx >> 5) & 127, k = idx & 31;
            eWT[idx] = (_Float16)((k < Dd) ? eW[(l * Dd + k) * Hh + col] : 0.f);
        }
        break;
    case 2: // m1wT [2][128][128]
        if (idx < 2 * 128 * 128) {
            int l = idx >> 14, col = (idx >> 7) & 127, k = idx & 127;
            m1wT[idx] = (_Float16)m1w[(l * Hh + k) * Hh + col];
        }
        break;
    case 3: // m2wT [2][128][128]
        if (idx < 2 * 128 * 128) {
            int l = idx >> 14, col = (idx >> 7) & 127, k = idx & 127;
            m2wT[idx] = (_Float16)m2w[(l * Hh + k) * Hh + col];
        }
        break;
    case 4: // wihT [384][128]
        if (idx < 384 * 128) {
            int col = idx >> 7, k = idx & 127;
            wihT[idx] = (_Float16)wih[k * 384 + col];
        }
        break;
    case 5: // whhT [384][128]
        if (idx < 384 * 128) {
            int col = idx >> 7, k = idx & 127;
            whhT[idx] = (_Float16)whh[k * 384 + col];
        }
        break;
    }
}

// ---------------------------------------------------------------------------
// Fragment loaders (CDNA5 ISA 7.12.2 layouts).
// 16-bit A 16x32: lane<16 -> row=lane, halves 0..7 = K(k0..k0+7), 8..15 = K(k0+16..+23)
//                 lane>=16 -> row=lane-16, K offset +8 in each half.
// 16-bit B 32x16: lane<16 -> col=lane, same K layout as A (0..7 = K(k0..k0+7), 8..15 = K(k0+16..+23)); lane>=16 -> K offset +8.
// ---------------------------------------------------------------------------
template <bool AF32, bool KHALF, int LDA>
__device__ inline v16h load_a_frag(const void* __restrict__ Asrc,
                                   int row0, int k0, int r, int hi) {
    v16h a;
    if (AF32) {
        const float* ap = (const float*)Asrc + (size_t)(row0 + r) * LDA + k0 + hi * 8;
        float4 q0 = *(const float4*)(ap);
        float4 q1 = *(const float4*)(ap + 4);
        a[0] = (_Float16)q0.x; a[1] = (_Float16)q0.y; a[2] = (_Float16)q0.z; a[3] = (_Float16)q0.w;
        a[4] = (_Float16)q1.x; a[5] = (_Float16)q1.y; a[6] = (_Float16)q1.z; a[7] = (_Float16)q1.w;
        if (KHALF) {
#pragma unroll
            for (int i = 0; i < 8; ++i) a[8 + i] = (_Float16)0.f;
        } else {
            float4 q2 = *(const float4*)(ap + 16);
            float4 q3 = *(const float4*)(ap + 20);
            a[8]  = (_Float16)q2.x; a[9]  = (_Float16)q2.y; a[10] = (_Float16)q2.z; a[11] = (_Float16)q2.w;
            a[12] = (_Float16)q3.x; a[13] = (_Float16)q3.y; a[14] = (_Float16)q3.z; a[15] = (_Float16)q3.w;
        }
    } else {
        const _Float16* ap = (const _Float16*)Asrc + (size_t)(row0 + r) * LDA + k0 + hi * 8;
        v8h q0 = *(const v8h*)(ap);
#pragma unroll
        for (int i = 0; i < 8; ++i) a[i] = q0[i];
        if (KHALF) {
#pragma unroll
            for (int i = 0; i < 8; ++i) a[8 + i] = (_Float16)0.f;
        } else {
            v8h q1 = *(const v8h*)(ap + 16);
#pragma unroll
            for (int i = 0; i < 8; ++i) a[8 + i] = q1[i];
        }
    }
    return a;
}

template <int LDB>
__device__ inline v16h load_b_frag(const _Float16* __restrict__ Bt,
                                   int col0, int k0, int r, int hi) {
    const _Float16* bp = Bt + (size_t)(col0 + r) * LDB + k0 + hi * 8;
    v8h b0 = *(const v8h*)(bp);
    v8h b1 = *(const v8h*)(bp + 16);
    v16h b;
#pragma unroll
    for (int i = 0; i < 8; ++i) { b[i] = b0[i]; b[8 + i] = b1[i]; }
    return b;
}

// ---------------------------------------------------------------------------
// Specialized WMMA GEMM: D[M x Ncols] = A[M x K] * Bt^T + bias.
// Each wave computes a 16x32 strip (one A fragment, two B fragments, two
// v_wmma_f32_16x16x32_f16 per K step). KSTEPS fully unrolled; all leading
// dims are template constants so stores fold into immediate offsets.
//   MODE: 0 = store f32, 1 = store f16, 2 = relu -> store f16,
//         3 = store f32 to Dst AND Dst2 (encoder: x and agg).
// ---------------------------------------------------------------------------
template <bool AF32, bool KHALF, int MODE, int KSTEPS, int LDA, int LDB, int LDD>
__global__ void __launch_bounds__(BDIM) wmma_gemm(
    const void* __restrict__ Asrc,
    const _Float16* __restrict__ Bt,
    const float* __restrict__ bias,
    void* __restrict__ Dst, void* __restrict__ Dst2,
    int Mtiles, int NPairs) {
    int wid = blockIdx.x * (BDIM / 32) + (threadIdx.x >> 5);
    if (wid >= Mtiles * NPairs) return;
    int tN = wid % NPairs, tM = wid / NPairs;
    int lane = threadIdx.x & 31;
    int r    = lane & 15;       // A-row / B-col within tile
    int hi   = lane >> 4;       // lane-half selects K sub-block

    int row0 = tM * 16, col0 = tN * 32;

    v8f c0, c1;
#pragma unroll
    for (int i = 0; i < 8; ++i) { c0[i] = 0.f; c1[i] = 0.f; }

#pragma unroll
    for (int ks = 0; ks < KSTEPS; ++ks) {
        int k0 = ks * 32;
        v16h a  = load_a_frag<AF32, KHALF, LDA>(Asrc, row0, k0, r, hi);
        v16h b0 = load_b_frag<LDB>(Bt, col0,      k0, r, hi);
        v16h b1 = load_b_frag<LDB>(Bt, col0 + 16, k0, r, hi);
        c0 = __builtin_amdgcn_wmma_f32_16x16x32_f16(false, a, false, b0, (short)0, c0, false, false);
        c1 = __builtin_amdgcn_wmma_f32_16x16x32_f16(false, a, false, b1, (short)0, c1, false, false);
    }

    int rbase = row0 + hi * 8;
    float bv0 = bias[col0 + r];
    float bv1 = bias[col0 + 16 + r];
    if (MODE == 0 || MODE == 3) {
        float* dp = (float*)Dst + (size_t)rbase * LDD + col0 + r;
#pragma unroll
        for (int v = 0; v < 8; ++v) {
            dp[v * LDD]      = c0[v] + bv0;
            dp[v * LDD + 16] = c1[v] + bv1;
        }
        if (MODE == 3) {
            float* dq = (float*)Dst2 + (size_t)rbase * LDD + col0 + r;
#pragma unroll
            for (int v = 0; v < 8; ++v) {
                dq[v * LDD]      = c0[v] + bv0;
                dq[v * LDD + 16] = c1[v] + bv1;
            }
        }
    } else {
        _Float16* dp = (_Float16*)Dst + (size_t)rbase * LDD + col0 + r;
#pragma unroll
        for (int v = 0; v < 8; ++v) {
            float v0 = c0[v] + bv0;
            float v1 = c1[v] + bv1;
            if (MODE == 2) {
                v0 = v0 > 0.f ? v0 : 0.f;
                v1 = v1 > 0.f ? v1 : 0.f;
            }
            dp[v * LDD]      = (_Float16)v0;
            dp[v * LDD + 16] = (_Float16)v1;
        }
    }
}

__global__ void zero_f32v4(float4* __restrict__ dst, int n4) {
    int i = blockIdx.x * blockDim.x + threadIdx.x;
    if (i < n4) dst[i] = make_float4(0.f, 0.f, 0.f, 0.f);
}

// ---------------------------------------------------------------------------
// Message pass: msg = relu(x[src] + e_emb); agg[dst] += msg (f32 L2 atomics).
// One thread per (edge, 4 channels); both batches share one e_emb read so the
// only HBM stream (e_emb, f16) is read once per (t,l). Prefetch ~16KB ahead
// on that stream (global_prefetch_b8).
// ---------------------------------------------------------------------------
__global__ void __launch_bounds__(BDIM) message_kernel(
    const int* __restrict__ src, const int* __restrict__ dst,
    const _Float16* __restrict__ eemb, const float* __restrict__ x,
    float* __restrict__ agg) {
    int idx = blockIdx.x * blockDim.x + threadIdx.x;
    int e = idx >> 5;
    if (e >= Ee) return;
    int hc = (idx & 31) << 2;
    int s = src[e], d = dst[e];
    const _Float16* ep = eemb + (size_t)e * Hh + hc;
    __builtin_prefetch(ep + 64 * Hh, 0, 1);   // stream 64 edges (16KB) ahead
    float e0 = (float)ep[0], e1 = (float)ep[1], e2 = (float)ep[2], e3 = (float)ep[3];
#pragma unroll
    for (int b = 0; b < Bsz; ++b) {
        float4 xv = *(const float4*)(x + ((size_t)b * Nn + s) * Hh + hc);
        float m0 = xv.x + e0; m0 = m0 > 0.f ? m0 : 0.f;
        float m1 = xv.y + e1; m1 = m1 > 0.f ? m1 : 0.f;
        float m2 = xv.z + e2; m2 = m2 > 0.f ? m2 : 0.f;
        float m3 = xv.w + e3; m3 = m3 > 0.f ? m3 : 0.f;
        float* ap = agg + ((size_t)b * Nn + d) * Hh + hc;
        atomicAdd(ap + 0, m0);
        atomicAdd(ap + 1, m1);
        atomicAdd(ap + 2, m2);
        atomicAdd(ap + 3, m3);
    }
}

// ---------------------------------------------------------------------------
// LayerNorm over H=128 + relu + residual; one wave32 per row (4 ch/lane).
// Writes x (residual stream) and agg (next layer's scatter base); optionally
// also emits the f16 copy consumed by the GRU GEMM (last layer).
// ---------------------------------------------------------------------------
__global__ void __launch_bounds__(BDIM) ln_relu_res(
    const float* __restrict__ h2, const float* __restrict__ g, const float* __restrict__ bb,
    float* __restrict__ x, float* __restrict__ agg, _Float16* __restrict__ sp_t, int rows) {
    int lane = threadIdx.x & 31;
    int row = blockIdx.x * (BDIM / 32) + (threadIdx.x >> 5);
    if (row >= rows) return;
    const float* hp = h2 + (size_t)row * Hh;
    float4 v = *(const float4*)(hp + lane * 4);
    float s  = v.x + v.y + v.z + v.w;
    float sq = v.x * v.x + v.y * v.y + v.z * v.z + v.w * v.w;
#pragma unroll
    for (int m = 16; m >= 1; m >>= 1) {
        s  += __shfl_xor(s, m, 32);
        sq += __shfl_xor(sq, m, 32);
    }
    float mu  = s * (1.f / 128.f);
    float var = sq * (1.f / 128.f) - mu * mu;
    float inv = rsqrtf(var + 1e-5f);
    float* xp = x + (size_t)row * Hh;
    float4 res = *(const float4*)(xp + lane * 4);
    float4 gg = *(const float4*)(g + lane * 4);
    float4 bv = *(const float4*)(bb + lane * 4);
    float y0 = (v.x - mu) * inv * gg.x + bv.x; y0 = y0 > 0.f ? y0 : 0.f;
    float y1 = (v.y - mu) * inv * gg.y + bv.y; y1 = y1 > 0.f ? y1 : 0.f;
    float y2 = (v.z - mu) * inv * gg.z + bv.z; y2 = y2 > 0.f ? y2 : 0.f;
    float y3 = (v.w - mu) * inv * gg.w + bv.w; y3 = y3 > 0.f ? y3 : 0.f;
    float4 o = make_float4(y0 + res.x, y1 + res.y, y2 + res.z, y3 + res.w);
    *(float4*)(xp + lane * 4) = o;
    *(float4*)(agg + (size_t)row * Hh + lane * 4) = o;
    if (sp_t) {
        _Float16* spp = sp_t + (size_t)row * Hh + lane * 4;
        spp[0] = (_Float16)o.x; spp[1] = (_Float16)o.y;
        spp[2] = (_Float16)o.z; spp[3] = (_Float16)o.w;
    }
}

// ---------------------------------------------------------------------------
// GRU gate fusion (torch order r,z,n): h' = (1-z)*n + z*h, in-place on h.
// Also emits f16 h for the next step's recurrent GEMM (f16 A path).
// ---------------------------------------------------------------------------
__global__ void __launch_bounds__(BDIM) gru_gate(
    const float* __restrict__ gi, const float* __restrict__ gh,
    float* __restrict__ h, _Float16* __restrict__ h16) {
    int idx = blockIdx.x * blockDim.x + threadIdx.x;
    if (idx >= BN * Hh) return;
    int m = idx >> 7, j = idx & 127;
    const float* gip = gi + (size_t)m * 384;
    const float* ghp = gh + (size_t)m * 384;
    float r = 1.f / (1.f + __expf(-(gip[j] + ghp[j])));
    float z = 1.f / (1.f + __expf(-(gip[128 + j] + ghp[128 + j])));
    float n = tanhf(gip[256 + j] + r * ghp[256 + j]);
    float hv = (1.f - z) * n + z * h[idx];
    h[idx] = hv;
    h16[idx] = (_Float16)hv;
}

// ---------------------------------------------------------------------------
// Head: pred = h @ head_w + head_b ; one wave32 per row, shfl reduction.
// ---------------------------------------------------------------------------
__global__ void __launch_bounds__(BDIM) head_kernel(
    const float* __restrict__ h, const float* __restrict__ w,
    const float* __restrict__ b, float* __restrict__ out) {
    int lane = threadIdx.x & 31;
    int row = blockIdx.x * (BDIM / 32) + (threadIdx.x >> 5);
    if (row >= BN) return;
    const float* hp = h + (size_t)row * Hh;
    float s = 0.f;
#pragma unroll
    for (int i = 0; i < 4; ++i) s += hp[lane + 32 * i] * w[lane + 32 * i];
#pragma unroll
    for (int m = 16; m >= 1; m >>= 1) s += __shfl_xor(s, m, 32);
    if (lane == 0) out[row] = s + b[0];
}

// ---------------------------------------------------------------------------
extern "C" void kernel_launch(void* const* d_in, const int* in_sizes, int n_in,
                              void* d_out, int out_size, void* d_ws, size_t ws_size,
                              hipStream_t stream) {
    (void)in_sizes; (void)n_in; (void)out_size; (void)ws_size;
    const float* x_seq     = (const float*)d_in[0];
    const int*   edge_idx  = (const int*)d_in[1];
    const float* edge_attr = (const float*)d_in[2];
    const float* enc_w  = (const float*)d_in[3];
    const float* enc_b  = (const float*)d_in[4];
    const float* eW     = (const float*)d_in[5];
    const float* eb     = (const float*)d_in[6];
    const float* m1w    = (const float*)d_in[7];
    const float* m1b    = (const float*)d_in[8];
    const float* m2w    = (const float*)d_in[9];
    const float* m2b    = (const float*)d_in[10];
    const float* ln_g   = (const float*)d_in[11];
    const float* ln_b   = (const float*)d_in[12];
    const float* wih    = (const float*)d_in[13];
    const float* whh    = (const float*)d_in[14];
    const float* bih    = (const float*)d_in[15];
    const float* bhh    = (const float*)d_in[16];
    const float* head_w = (const float*)d_in[17];
    const float* head_b = (const float*)d_in[18];
    float* out = (float*)d_out;

    const int* src = edge_idx;
    const int* dst = edge_idx + Ee;

    // Workspace carve-out (256B aligned)
    char* ws = (char*)d_ws;
    size_t off = 0;
    auto alloc = [&](size_t bytes) -> void* {
        off = (off + 255) & ~(size_t)255;
        void* p = ws + off;
        off += bytes;
        return p;
    };
    _Float16* e_emb  = (_Float16*)alloc((size_t)Ll * Ee * Hh * 2);   // 164 MB, the only HBM stream
    _Float16* enc_wT = (_Float16*)alloc(128 * 32 * 2);
    _Float16* eWT    = (_Float16*)alloc(2 * 128 * 32 * 2);
    _Float16* m1wT   = (_Float16*)alloc(2 * 128 * 128 * 2);
    _Float16* m2wT   = (_Float16*)alloc(2 * 128 * 128 * 2);
    _Float16* wihT   = (_Float16*)alloc(384 * 128 * 2);
    _Float16* whhT   = (_Float16*)alloc(384 * 128 * 2);
    float*    xbuf   = (float*)alloc((size_t)BN * Hh * 4);
    float*    aggbuf = (float*)alloc((size_t)BN * Hh * 4);
    _Float16* h1buf  = (_Float16*)alloc((size_t)BN * Hh * 2);
    float*    h2buf  = (float*)alloc((size_t)BN * Hh * 4);
    _Float16* spbuf  = (_Float16*)alloc((size_t)Tt * BN * Hh * 2);
    float*    hstate = (float*)alloc((size_t)BN * Hh * 4);
    _Float16* h16buf = (_Float16*)alloc((size_t)BN * Hh * 2);
    float*    gibuf  = (float*)alloc((size_t)BN * 384 * 4);
    float*    ghbuf  = (float*)alloc((size_t)BN * 384 * 4);

    // 1) Pack weights to f16 transposed WMMA-B layout
    pack_weights<<<dim3(192, 6), BDIM, 0, stream>>>(enc_w, eW, m1w, m2w, wih, whh,
                                                    enc_wT, eWT, m1wT, m2wT, wihT, whhT);

    // 2) Time-invariant edge embeddings: e_emb[l] = edge_attr @ eW[l] + eb[l] (f16)
    for (int l = 0; l < Ll; ++l) {
        int waves = (Ee / 16) * 4;
        wmma_gemm<true, true, 1, 1, Ff, 32, Hh><<<(waves + 7) / 8, BDIM, 0, stream>>>(
            edge_attr, eWT + (size_t)l * 128 * 32, eb + l * Hh,
            e_emb + (size_t)l * Ee * Hh, nullptr, Ee / 16, 4);
    }

    // 3) Spatial stack over T
    for (int t = 0; t < Tt; ++t) {
        // encoder: x = x_seq[:,t] @ enc_w + enc_b ; dual-store to x and agg
        for (int b = 0; b < Bsz; ++b) {
            int waves = (Nn / 16) * 4;
            wmma_gemm<true, true, 3, 1, Ff, 32, Hh><<<(waves + 7) / 8, BDIM, 0, stream>>>(
                x_seq + ((size_t)(b * Tt + t)) * Nn * Ff, enc_wT, enc_b,
                xbuf + (size_t)b * Nn * Hh, aggbuf + (size_t)b * Nn * Hh, Nn / 16, 4);
        }
        for (int l = 0; l < Ll; ++l) {
            // scatter-add relu(x[src]+e_emb) onto agg (pre-seeded with x)
            message_kernel<<<(Ee * 32 + BDIM - 1) / BDIM, BDIM, 0, stream>>>(
                src, dst, e_emb + (size_t)l * Ee * Hh, xbuf, aggbuf);
            // h1 = relu(agg @ m1w + m1b)  -> f16
            {
                int waves = (BN / 16) * 4;
                wmma_gemm<true, false, 2, 4, Hh, Hh, Hh><<<(waves + 7) / 8, BDIM, 0, stream>>>(
                    aggbuf, m1wT + (size_t)l * Hh * Hh, m1b + l * Hh,
                    h1buf, nullptr, BN / 16, 4);
            }
            // h2 = h1 @ m2w + m2b -> f32
            {
                int waves = (BN / 16) * 4;
                wmma_gemm<false, false, 0, 4, Hh, Hh, Hh><<<(waves + 7) / 8, BDIM, 0, stream>>>(
                    h1buf, m2wT + (size_t)l * Hh * Hh, m2b + l * Hh,
                    h2buf, nullptr, BN / 16, 4);
            }
            // x = relu(LN(h2)) + x ; also re-seed agg; last layer emits f16 seq
            ln_relu_res<<<(BN + 7) / 8, BDIM, 0, stream>>>(
                h2buf, ln_g + l * Hh, ln_b + l * Hh, xbuf, aggbuf,
                (l == Ll - 1) ? (spbuf + (size_t)t * BN * Hh) : (_Float16*)nullptr, BN);
        }
    }

    // 4) GRU over T (h kept in f32 for the gate update, f16 for the GEMM A path)
    zero_f32v4<<<(BN * Hh / 4 + BDIM - 1) / BDIM, BDIM, 0, stream>>>((float4*)hstate, BN * Hh / 4);
    zero_f32v4<<<(BN * Hh * 2 / 16 + BDIM - 1) / BDIM, BDIM, 0, stream>>>((float4*)h16buf, BN * Hh * 2 / 16);
    for (int t = 0; t < Tt; ++t) {
        int waves = (BN / 16) * 12;   // 384 cols -> 12 N-pairs
        wmma_gemm<false, false, 0, 4, Hh, Hh, 384><<<(waves + 7) / 8, BDIM, 0, stream>>>(
            spbuf + (size_t)t * BN * Hh, wihT, bih, gibuf, nullptr, BN / 16, 12);
        wmma_gemm<false, false, 0, 4, Hh, Hh, 384><<<(waves + 7) / 8, BDIM, 0, stream>>>(
            h16buf, whhT, bhh, ghbuf, nullptr, BN / 16, 12);
        gru_gate<<<(BN * Hh + BDIM - 1) / BDIM, BDIM, 0, stream>>>(gibuf, ghbuf, hstate, h16buf);
    }

    // 5) Head
    head_kernel<<<(BN + 7) / 8, BDIM, 0, stream>>>(hstate, head_w, head_b, out);
}
